// GraphSAGEModel_18098992185492
// MI455X (gfx1250) — matrix operand order repaired and checked by reading, not compiled, not run on hardware
//
#include <hip/hip_runtime.h>

#define N_NODES 50000
#define N_EDGES 800000
#define M_TILES (N_NODES / 16)   // 3125

typedef __attribute__((ext_vector_type(16))) __bf16        v16bf;
typedef __attribute__((ext_vector_type(8)))  float         v8f;
typedef __attribute__((ext_vector_type(8)))  unsigned int  v8u;
typedef __attribute__((ext_vector_type(4)))  unsigned int  v4u;
typedef __attribute__((ext_vector_type(8)))  int           i32x8;
typedef __attribute__((ext_vector_type(4)))  int           i32x4;

#if __has_builtin(__builtin_amdgcn_tensor_load_to_lds) && \
    __has_builtin(__builtin_amdgcn_s_wait_tensorcnt)
#define HAVE_TDM 1
#else
#define HAVE_TDM 0
#endif

// pack two f32 -> packed bf16 pair: +0x8000 round, then v_perm_b32 grabs high halves
__device__ __forceinline__ unsigned int pk_bf16(float lo, float hi) {
    return __builtin_amdgcn_perm(__float_as_uint(hi) + 0x8000u,
                                 __float_as_uint(lo) + 0x8000u, 0x07060302u);
}
__device__ __forceinline__ unsigned short bf16_1(float v) {
    return (unsigned short)((__float_as_uint(v) + 0x8000u) >> 16);
}

// B fragment fetch from LDS: 8 bf16 at Bs[idx] and 8 bf16 at Bs[idx+16] (two ds_load_b128)
__device__ __forceinline__ v16bf ldB(const unsigned short* Bs, int idx) {
    const v4u* bp = (const v4u*)(Bs + idx);
    v4u q0 = bp[0], q1 = bp[2];
    v8u bu;
    bu[0] = q0[0]; bu[1] = q0[1]; bu[2] = q0[2]; bu[3] = q0[3];
    bu[4] = q1[0]; bu[5] = q1[1]; bu[6] = q1[2]; bu[7] = q1[3];
    return __builtin_bit_cast(v16bf, bu);
}

// ---------- utility kernels ----------
__global__ void zero_f4(float4* p, int n4) {
    int i = blockIdx.x * blockDim.x + threadIdx.x;
    if (i < n4) p[i] = make_float4(0.f, 0.f, 0.f, 0.f);
}
__global__ void zero_i32(int* p, int n) {
    int i = blockIdx.x * blockDim.x + threadIdx.x;
    if (i < n) p[i] = 0;
}
__global__ void f32_to_bf16(const float* __restrict__ src, unsigned short* __restrict__ dst, int n2) {
    int i = blockIdx.x * blockDim.x + threadIdx.x;
    if (i >= n2) return;
    float2 v = ((const float2*)src)[i];
    ((unsigned int*)dst)[i] = pk_bf16(v.x, v.y);
}
__global__ void deg_kernel(const long long* __restrict__ ei, int* __restrict__ deg) {
    int e = blockIdx.x * blockDim.x + threadIdx.x;
    if (e >= N_EDGES) return;
    atomicAdd(&deg[(int)ei[e + N_EDGES]], 1);
}
__global__ void invdeg_kernel(const int* __restrict__ deg, float* __restrict__ inv) {
    int i = blockIdx.x * blockDim.x + threadIdx.x;
    if (i >= N_NODES) return;
    int d = deg[i];
    inv[i] = 1.0f / (float)(d > 1 ? d : 1);
}

// Combined transposed bf16 weight: Wt[col*256 + k]; k<128 from Wl, k>=128 from Wr.
__global__ void prep_w(const float* __restrict__ Wl, const float* __restrict__ Wr,
                       unsigned short* __restrict__ Wt, int ncols) {
    int idx = blockIdx.x * blockDim.x + threadIdx.x;
    if (idx >= ncols * 256) return;
    int col = idx >> 8, k = idx & 255;
    float v = (k < 128) ? Wl[k * ncols + col] : Wr[(k - 128) * ncols + col];
    Wt[col * 256 + k] = bf16_1(v);
}

// Scatter-add: one wave per edge; 32 lanes x 4 bf16 feats (uint2 gather), f32 HW atomics.
__global__ void scatter_bf(const unsigned short* __restrict__ h,
                           const long long* __restrict__ ei,
                           float* __restrict__ agg) {
    int e    = blockIdx.x * (blockDim.x >> 5) + (threadIdx.x >> 5);
    int lane = threadIdx.x & 31;
    if (e >= N_EDGES) return;
    long long s = ei[e];
    long long d = ei[e + N_EDGES];
    uint2 w = ((const uint2*)(h + (size_t)s * 128))[lane];
    float* ap = agg + (size_t)d * 128 + lane * 4;
    unsafeAtomicAdd(ap + 0, __uint_as_float(w.x << 16));
    unsafeAtomicAdd(ap + 1, __uint_as_float(w.x & 0xFFFF0000u));
    unsafeAtomicAdd(ap + 2, __uint_as_float(w.y << 16));
    unsafeAtomicAdd(ap + 3, __uint_as_float(w.y & 0xFFFF0000u));
}

// ---------- fused SAGE GEMM: out = relu?( [act_bf | agg*inv] @ Wt^T + b ) ----------
// One wave per 16-row tile; NT column tiles of 16. K=256 (bf16 self | f32 mean-agg).
// Weights staged into LDS once per block via the Tensor Data Mover (TDM).
template <int NT, bool RELU, bool OUT_BF16>
__global__ void sage_gemm(const unsigned short* __restrict__ act_bf,
                          const float* __restrict__ agg,
                          const float* __restrict__ invdeg,
                          const unsigned short* __restrict__ Wt,
                          const float* __restrict__ bias,
                          void* __restrict__ out) {
    __shared__ alignas(16) unsigned short Bs[NT * 16 * 256];

#if HAVE_TDM
    // One TDM descriptor: 1 row of NT*16*256*2 bytes (data_size=8B units), LDS offset 0.
    if (threadIdx.x < 32) {                       // one wave issues the DMA
        const unsigned long long ga = (unsigned long long)Wt;
        const unsigned int n8 = (unsigned int)(NT * 16 * 256 * 2 / 8);  // 8-byte elems
        v4u g0;
        g0[0] = 1u;                                             // count=1 (valid user D#)
        g0[1] = 0u;                                             // lds_addr = 0 (Bs base)
        g0[2] = (unsigned int)(ga & 0xFFFFFFFFu);               // global_addr[31:0]
        g0[3] = (unsigned int)((ga >> 32) & 0x01FFFFFFu)        // global_addr[56:32]
              | (2u << 30);                                     // type = 2 ("image")
        i32x8 g1;
        g1[0] = (int)(3u << 16);            // workgroup_mask=0, data_size=3 (8B)
        g1[1] = (int)((n8 & 0xFFFFu) << 16);// tensor_dim0[15:0]   (barrier addr = 0)
        g1[2] = (int)((n8 >> 16) | (1u << 16)); // tensor_dim0[31:16] | tensor_dim1=1
        g1[3] = (int)((n8 & 0xFFFFu) << 16);// tile_dim0 = n8
        g1[4] = 1;                          // tile_dim1 = 1, tile_dim2 = 0
        g1[5] = (int)n8;                    // tensor_dim0_stride[31:0]
        g1[6] = 0; g1[7] = 0;               // stride hi, tensor_dim1_stride = 0
        i32x4 z4 = {0, 0, 0, 0};
#if __clang_major__ >= 23
        i32x8 z8 = {0, 0, 0, 0, 0, 0, 0, 0};
        __builtin_amdgcn_tensor_load_to_lds(g0, g1, z4, z4, z8, 0);
#else
        __builtin_amdgcn_tensor_load_to_lds(g0, g1, z4, z4, 0);
#endif
        __builtin_amdgcn_s_wait_tensorcnt(0);
    }
#else
    for (int i = threadIdx.x; i < NT * 512; i += 256)
        ((v4u*)Bs)[i] = ((const v4u*)Wt)[i];
#endif
    __syncthreads();

    const int wave = threadIdx.x >> 5;
    const int lane = threadIdx.x & 31;
    const int tile = blockIdx.x * (blockDim.x >> 5) + wave;
    if (tile >= M_TILES) return;                 // wave-uniform: EXEC all-1s for WMMA

    const int  m0   = tile * 16;
    const int  rloc = lane & 15;
    const bool hi   = lane >= 16;
    const int  row  = m0 + rloc;                 // A-matrix row this lane feeds
    const float inv = invdeg[row];

    v8f acc[NT];
#pragma unroll
    for (int nt = 0; nt < NT; ++nt)
#pragma unroll
        for (int r = 0; r < 8; ++r) acc[nt][r] = 0.0f;

    // ---- K = 0..127 : self features, already bf16 -> pure loads, no conversion ----
#pragma unroll
    for (int kb = 0; kb < 4; ++kb) {
        const int kbase = kb * 32 + (hi ? 8 : 0);
        const v4u* ap = (const v4u*)(act_bf + (size_t)row * 128 + kbase);
        v4u a0 = ap[0];          // K = kbase .. kbase+7
        v4u a1 = ap[2];          // K = kbase+16 .. kbase+23
        v8u au;
        au[0] = a0[0]; au[1] = a0[1]; au[2] = a0[2]; au[3] = a0[3];
        au[4] = a1[0]; au[5] = a1[1]; au[6] = a1[2]; au[7] = a1[3];
        v16bf a = __builtin_bit_cast(v16bf, au);

        const int kstart = kb * 32 + (hi ? 8 : 0);
        v16bf bcur = ldB(Bs, rloc * 256 + kstart);           // nt = 0
#pragma unroll
        for (int nt = 0; nt < NT; ++nt) {
            v16bf bnxt;
            if (nt + 1 < NT) bnxt = ldB(Bs, ((nt + 1) * 16 + rloc) * 256 + kstart);
            acc[nt] = __builtin_amdgcn_wmma_f32_16x16x32_bf16(
                false, a, false, bcur, (short)0, acc[nt], false, false);
            if (nt + 1 < NT) bcur = bnxt;
        }
    }

    // ---- K = 128..255 : mean-aggregated neighbors (fp32 * inv_deg -> bf16) ----
#pragma unroll
    for (int kb = 4; kb < 8; ++kb) {
        const int kbase = kb * 32 + (hi ? 8 : 0) - 128;
        const float4* p0 = (const float4*)(agg + (size_t)row * 128 + kbase);
        float4 f0 = p0[0], f1 = p0[1], f2 = p0[4], f3 = p0[5];
        v8u au;
        au[0] = pk_bf16(f0.x * inv, f0.y * inv); au[1] = pk_bf16(f0.z * inv, f0.w * inv);
        au[2] = pk_bf16(f1.x * inv, f1.y * inv); au[3] = pk_bf16(f1.z * inv, f1.w * inv);
        au[4] = pk_bf16(f2.x * inv, f2.y * inv); au[5] = pk_bf16(f2.z * inv, f2.w * inv);
        au[6] = pk_bf16(f3.x * inv, f3.y * inv); au[7] = pk_bf16(f3.z * inv, f3.w * inv);
        v16bf a = __builtin_bit_cast(v16bf, au);

        const int kstart = kb * 32 + (hi ? 8 : 0);
        v16bf bcur = ldB(Bs, rloc * 256 + kstart);           // nt = 0
#pragma unroll
        for (int nt = 0; nt < NT; ++nt) {
            v16bf bnxt;
            if (nt + 1 < NT) bnxt = ldB(Bs, ((nt + 1) * 16 + rloc) * 256 + kstart);
            acc[nt] = __builtin_amdgcn_wmma_f32_16x16x32_bf16(
                false, a, false, bcur, (short)0, acc[nt], false, false);
            if (nt + 1 < NT) bcur = bnxt;
        }
    }

    // ---- epilogue: D layout -> lane L: col = L&15, rows r + (L>=16 ? 8 : 0) ----
    const int NCOLS = NT * 16;
#pragma unroll
    for (int nt = 0; nt < NT; ++nt) {
        const int  col = nt * 16 + rloc;
        const float bb = bias[col];
#pragma unroll
        for (int r = 0; r < 8; ++r) {
            float v = acc[nt][r] + bb;
            if (RELU) v = v > 0.f ? v : 0.f;
            const int orow = m0 + r + (hi ? 8 : 0);
            if (OUT_BF16)
                ((unsigned short*)out)[(size_t)orow * NCOLS + col] = bf16_1(v);
            else
                ((float*)out)[(size_t)orow * NCOLS + col] = v;
        }
    }
}

// ---------- host launch ----------
extern "C" void kernel_launch(void* const* d_in, const int* in_sizes, int n_in,
                              void* d_out, int out_size, void* d_ws, size_t ws_size,
                              hipStream_t stream) {
    (void)in_sizes; (void)n_in; (void)out_size; (void)ws_size;

    const float*     x   = (const float*)d_in[0];
    const long long* ei  = (const long long*)d_in[1];   // int64 [2, N_EDGES]
    const float*     Wl1 = (const float*)d_in[2];
    const float*     Wr1 = (const float*)d_in[3];
    const float*     b1  = (const float*)d_in[4];
    const float*     Wl2 = (const float*)d_in[5];
    const float*     Wr2 = (const float*)d_in[6];
    const float*     b2  = (const float*)d_in[7];
    const float*     Wl3 = (const float*)d_in[8];
    const float*     Wr3 = (const float*)d_in[9];
    const float*     b3  = (const float*)d_in[10];
    float* out = (float*)d_out;

    char* ws = (char*)d_ws;
    const size_t AGG_BYTES = (size_t)N_NODES * 128 * sizeof(float);          // 25.6 MB
    const size_t BF_BYTES  = (size_t)N_NODES * 128 * sizeof(unsigned short); // 12.8 MB
    float*          agg = (float*)(ws);
    unsigned short* xbf = (unsigned short*)(ws + AGG_BYTES);
    unsigned short* h1  = (unsigned short*)(ws + AGG_BYTES + BF_BYTES);
    unsigned short* h2  = (unsigned short*)(ws + AGG_BYTES + 2 * BF_BYTES);
    int*            deg = (int*)  (ws + AGG_BYTES + 3 * BF_BYTES);
    float*          inv = (float*)(ws + AGG_BYTES + 3 * BF_BYTES + (size_t)N_NODES * 4);
    unsigned short* Wt1 = (unsigned short*)(ws + AGG_BYTES + 3 * BF_BYTES + (size_t)N_NODES * 8);
    unsigned short* Wt2 = Wt1 + 256 * 128;
    unsigned short* Wt3 = Wt2 + 256 * 128;

    // prep: x -> bf16, weights -> combined transposed bf16
    const int n2 = N_NODES * 128 / 2;
    f32_to_bf16<<<(n2 + 255) / 256, 256, 0, stream>>>(x, xbf, n2);
    prep_w<<<(256 * 128 + 255) / 256, 256, 0, stream>>>(Wl1, Wr1, Wt1, 128);
    prep_w<<<(256 * 128 + 255) / 256, 256, 0, stream>>>(Wl2, Wr2, Wt2, 128);
    prep_w<<<(256 * 64  + 255) / 256, 256, 0, stream>>>(Wl3, Wr3, Wt3, 64);

    // degrees (shared across layers)
    zero_i32<<<(N_NODES + 255) / 256, 256, 0, stream>>>(deg, N_NODES);
    deg_kernel<<<(N_EDGES + 255) / 256, 256, 0, stream>>>(ei, deg);
    invdeg_kernel<<<(N_NODES + 255) / 256, 256, 0, stream>>>(deg, inv);

    const int n4       = N_NODES * 128 / 4;
    const int zeroBlks = (n4 + 255) / 256;
    const int scatBlks = (N_EDGES + 7) / 8;   // 8 waves (edges) per 256-thread block
    const int gemmBlks = (M_TILES + 7) / 8;   // 8 waves (tiles) per 256-thread block

    // ---- layer 1: h1 = relu([x|agg] @ W1 + b1) ----
    zero_f4<<<zeroBlks, 256, 0, stream>>>((float4*)agg, n4);
    scatter_bf<<<scatBlks, 256, 0, stream>>>(xbf, ei, agg);
    sage_gemm<8, true, true><<<gemmBlks, 256, 0, stream>>>(xbf, agg, inv, Wt1, b1, h1);

    // ---- layer 2: h2 = relu([h1|agg] @ W2 + b2) ----
    zero_f4<<<zeroBlks, 256, 0, stream>>>((float4*)agg, n4);
    scatter_bf<<<scatBlks, 256, 0, stream>>>(h1, ei, agg);
    sage_gemm<8, true, true><<<gemmBlks, 256, 0, stream>>>(h1, agg, inv, Wt2, b2, h2);

    // ---- layer 3: out = [h2|agg] @ W3 + b3 (64 cols, fp32, no relu) ----
    zero_f4<<<zeroBlks, 256, 0, stream>>>((float4*)agg, n4);
    scatter_bf<<<scatBlks, 256, 0, stream>>>(h2, ei, agg);
    sage_gemm<4, false, false><<<gemmBlks, 256, 0, stream>>>(h2, agg, inv, Wt3, b3, out);
}
